// MoEAdapter_89945205113236
// MI455X (gfx1250) — matrix operand order repaired and checked by reading, not compiled, not run on hardware
//
#include <hip/hip_runtime.h>
#include <stdint.h>

// ---------- CDNA5 WMMA / TDM vector types ----------
typedef __attribute__((ext_vector_type(16))) __bf16       v16bf;
typedef __attribute__((ext_vector_type(8)))  float        v8f;
typedef __attribute__((ext_vector_type(4)))  unsigned int v4u;
typedef __attribute__((ext_vector_type(8)))  int          v8i;
typedef __attribute__((ext_vector_type(4)))  int          v4i;

#define ID_DIM  1024
#define LLM_DIM 4096
#define IN_DIM  5120
#define OUT_DIM 2048
#define HID     4096
#define NEXP    8

// round-to-nearest-even fp32 -> bf16 (raw bits)
__device__ __forceinline__ unsigned short f2bf(float f) {
  union { float f; unsigned int u; } c; c.f = f;
  unsigned int u = c.u;
  u += 0x7FFFu + ((u >> 16) & 1u);
  return (unsigned short)(u >> 16);
}

// ============================================================
// elementwise helpers
// ============================================================
__global__ void zero_f32_k(float* __restrict__ p, long long n4) {
  long long i = (long long)blockIdx.x * blockDim.x + threadIdx.x;
  if (i < n4) ((float4*)p)[i] = make_float4(0.f, 0.f, 0.f, 0.f);
}

__global__ void cvt_bf16_k(const float* __restrict__ src,
                           unsigned short* __restrict__ dst, long long n4) {
  long long i = (long long)blockIdx.x * blockDim.x + threadIdx.x;
  if (i >= n4) return;
  float4 v = ((const float4*)src)[i];
  unsigned int lo = (unsigned int)f2bf(v.x) | ((unsigned int)f2bf(v.y) << 16);
  unsigned int hi = (unsigned int)f2bf(v.z) | ((unsigned int)f2bf(v.w) << 16);
  ((uint2*)dst)[i] = make_uint2(lo, hi);
}

// concat(id_emb, llm_emb) -> bf16 [B x IN_DIM]
__global__ void pack_combined_k(const float* __restrict__ id_emb,
                                const float* __restrict__ llm_emb,
                                unsigned short* __restrict__ dst, int B) {
  long long i = (long long)blockIdx.x * blockDim.x + threadIdx.x; // float4 group
  long long tot = (long long)B * (IN_DIM / 4);
  if (i >= tot) return;
  int b  = (int)(i / (IN_DIM / 4));
  int c4 = (int)(i % (IN_DIM / 4)) * 4;
  float4 v = (c4 < ID_DIM)
      ? *(const float4*)(id_emb  + (long long)b * ID_DIM  + c4)
      : *(const float4*)(llm_emb + (long long)b * LLM_DIM + (c4 - ID_DIM));
  unsigned int lo = (unsigned int)f2bf(v.x) | ((unsigned int)f2bf(v.y) << 16);
  unsigned int hi = (unsigned int)f2bf(v.z) | ((unsigned int)f2bf(v.w) << 16);
  *(uint2*)(dst + (long long)b * IN_DIM + c4) = make_uint2(lo, hi);
}

// ============================================================
// gate: relu(x @ gW1 + gb1) @ gW2 + gb2 -> top2 softmax -> dense_w [B x 8]
// one wave32 per token (fp32 path, VALU + cross-lane reduce)
// ============================================================
__global__ __launch_bounds__(128) void gate_k(
    const float* __restrict__ id_emb, const float* __restrict__ llm_emb,
    const float* __restrict__ gW1, const float* __restrict__ gb1,
    const float* __restrict__ gW2, const float* __restrict__ gb2,
    float* __restrict__ dense_w, int B) {
  const int lane  = threadIdx.x & 31;
  const int wave  = threadIdx.x >> 5;
  const int token = blockIdx.x * 4 + wave;
  if (token >= B) return;

  float acc[16];
#pragma unroll
  for (int j = 0; j < 16; ++j) acc[j] = 0.f;

  for (int i = lane; i < IN_DIM; i += 32) {
    float x = (i < ID_DIM) ? id_emb[(long long)token * ID_DIM + i]
                           : llm_emb[(long long)token * LLM_DIM + (i - ID_DIM)];
    const float4* w = (const float4*)(gW1 + (long long)i * 16);
#pragma unroll
    for (int q = 0; q < 4; ++q) {
      float4 wv = w[q];
      acc[q * 4 + 0] += x * wv.x; acc[q * 4 + 1] += x * wv.y;
      acc[q * 4 + 2] += x * wv.z; acc[q * 4 + 3] += x * wv.w;
    }
  }
#pragma unroll
  for (int j = 0; j < 16; ++j) {
#pragma unroll
    for (int off = 16; off >= 1; off >>= 1)
      acc[j] += __shfl_xor(acc[j], off, 32);
    acc[j] = fmaxf(acc[j] + gb1[j], 0.f);      // hidden, identical on all lanes
  }
  float logit[8];
#pragma unroll
  for (int e = 0; e < 8; ++e) {
    float s = gb2[e];
#pragma unroll
    for (int j = 0; j < 16; ++j) s += acc[j] * gW2[j * 8 + e];
    logit[e] = s;
  }
  int i1 = 0; float l1 = logit[0];
#pragma unroll
  for (int e = 1; e < 8; ++e) if (logit[e] > l1) { l1 = logit[e]; i1 = e; }
  int i2 = -1; float l2 = -3.4e38f;
#pragma unroll
  for (int e = 0; e < 8; ++e)
    if (e != i1 && logit[e] > l2) { l2 = logit[e]; i2 = e; }
  float p2 = __expf(l2 - l1);
  float den = 1.f + p2;
  float wa = 1.f / den, wb = p2 / den;
  if (lane < 8)
    dense_w[(long long)token * NEXP + lane] =
        (lane == i1) ? wa : (lane == i2) ? wb : 0.f;
}

// ============================================================
// TDM: DMA one A tile (128 rows x 64B) global -> LDS, with LDS padding
// 16B per 64B row => LDS row stride 80B (matches PSTR=20 uints).
// Issued by one wave; tracked with TENSORcnt.
// Descriptor per CDNA5 ISA ch.8 (D# groups 0/1; 2D => trailing groups zero).
// This toolchain exposes the 6-arg builtin:
//   (uint32x4 g0, int32x8 g1, int32x4, int32x4, int32x8, i32 cpol)
// ============================================================
__device__ __forceinline__ void tdm_load_a_tile(unsigned lds_byte,
                                                const unsigned short* gptr,
                                                unsigned ldaDW) {
  unsigned long long ga = (unsigned long long)(size_t)gptr;
  v4u g0;
  g0[0] = 1u;                                        // count=1, no gather
  g0[1] = lds_byte;                                  // lds_addr
  g0[2] = (unsigned)(ga & 0xFFFFFFFFu);              // global_addr[31:0]
  g0[3] = (unsigned)((ga >> 32) & 0x01FFFFFFu)       // global_addr[56:32]
          | (2u << 30);                              // type=2 ("image")
  v8i g1;
  g1[0] = (int)((2u << 16)      // data_size = 4B
                | (1u << 20)    // pad_enable
                | (3u << 22)    // pad_interval: 16 DWORDs
                | (3u << 25));  // pad_amount:   4 DWORDs
  g1[1] = (int)(16u << 16);     // tensor_dim0[15:0]=16 DW (atomic_barrier=0)
  g1[2] = (int)(128u << 16);    // tensor_dim0[31:16]=0 | tensor_dim1[15:0]=128
  g1[3] = (int)(16u << 16);     // tensor_dim1[31:16]=0 | tile_dim0=16 DW
  g1[4] = (int)(128u);          // tile_dim1=128 | tile_dim2=0
  g1[5] = (int)(ldaDW);         // tensor_dim0_stride[31:0] (DWORD units)
  g1[6] = 0;                    // stride0[47:32]=0 | stride1[15:0]=0
  g1[7] = 0;                    // stride1[47:16]=0
  v4i z4 = {0, 0, 0, 0};
  v8i z8 = {0, 0, 0, 0, 0, 0, 0, 0};
  __builtin_amdgcn_tensor_load_to_lds(g0, g1, z4, z4, z8, 0);
}

// ============================================================
// bf16 WMMA GEMM: C(MxN) = A(MxK) * B(KxN), f32 accumulate
// EPI==0: store bf16 relu(acc + bias)        (expert layer 1 -> h)
// EPI==1: f32 C += rowScale * (acc + bias)   (expert layer 2 -> out)
//
// 256 threads (8 wave32), macro-tile 128x128, K-step 32.
// A tiles DMA'd by the Tensor Data Mover straight into padded LDS rows;
// B tiles staged by the waves with an in-register k-pair transpose.
// LDS holds packed bf16 k-pairs, so each WMMA fragment is two aligned
// ds_load_b128. Double-buffered, one barrier per K-step; next tile's
// TDM + global_load_b128s issue before the WMMAs.
// ============================================================
#define TM 128
#define TN 128
#define TK 32
#define PSTR 20       // padded pair-stride (uints): 80B rows, 16B-aligned

union AFrag { unsigned int u[8]; uint4 q[2]; v16bf v; };
union U4S   { uint4 q; unsigned short s[8]; };

template <int EPI>
__global__ __launch_bounds__(256) void gemm_bf16_wmma(
    const unsigned short* __restrict__ A, int lda,   // bf16 [M x K]
    const unsigned short* __restrict__ Bm, int ldb,  // bf16 [K x N]
    const float* __restrict__ bias,                  // [N]
    const float* __restrict__ scale, int sstride,    // per-row scale (EPI==1)
    void* __restrict__ Cvoid, int ldc, int K) {
  __shared__ __align__(16) unsigned int lAp[2][TM * PSTR];
  __shared__ __align__(16) unsigned int lBp[2][TN * PSTR];

  const int tid   = threadIdx.x;
  const int lane  = tid & 31;
  const int wave  = tid >> 5;
  const int waveM = wave >> 1;  // 0..3 -> 32-row slab
  const int waveN = wave & 1;   // 0..1 -> 64-col slab
  const int mBase = blockIdx.y * TM;
  const int nBase = blockIdx.x * TN;
  const int r16   = lane & 15;
  const int hi    = lane >> 4;  // 0/1: which half-wave

  v8f acc[2][4];
#pragma unroll
  for (int fm = 0; fm < 2; ++fm)
#pragma unroll
    for (int fn = 0; fn < 4; ++fn) acc[fm][fn] = {};

  // ---- B staging coordinates: thread -> (kpair=tid>>4, 8 cols at nn) ----
  const int bKp = tid >> 4;             // 0..15
  const int bNN = (tid & 15) * 8;

  const unsigned short* aTile = A + (long long)mBase * lda;  // TDM source rows
  const unsigned short* bP0 = Bm + (long long)(2 * bKp) * ldb + nBase + bNN;
  const unsigned short* bP1 = bP0 + ldb;
  const long long bStep = (long long)TK * ldb;
  const unsigned ldaDW = (unsigned)lda >> 1;
  const unsigned ldsA0 = (unsigned)(size_t)(void*)&lAp[0][0];
  const unsigned ldsA1 = (unsigned)(size_t)(void*)&lAp[1][0];

  uint4 bR0, bR1;
  auto gloadB = [&]() {
    bR0 = *(const uint4*)bP0;
    bR1 = *(const uint4*)bP1;
    bP0 += bStep; bP1 += bStep;
  };
  auto lstoreB = [&](int buf) {
    // interleave two k-rows into packed pairs, transpose to [n][kpair]
    U4S r0, r1; r0.q = bR0; r1.q = bR1;
    unsigned int* bp = &lBp[buf][bNN * PSTR + bKp];
#pragma unroll
    for (int e = 0; e < 8; ++e)
      bp[e * PSTR] = (unsigned int)r0.s[e] | ((unsigned int)r1.s[e] << 16);
  };

  const int nkt = K / TK;
  // ---- prologue: tile 0 ----
  if (wave == 0) tdm_load_a_tile(ldsA0, aTile, ldaDW);
  gloadB();
  lstoreB(0);
  if (wave == 0) __builtin_amdgcn_s_wait_tensorcnt(0);
  __syncthreads();

  for (int kt = 0; kt < nkt; ++kt) {
    const int buf = kt & 1;
    const bool more = (kt + 1 < nkt);
    if (more) {                           // overlap next tile with compute
      if (wave == 0)
        tdm_load_a_tile(buf ? ldsA0 : ldsA1, aTile + (kt + 1) * TK, ldaDW);
      gloadB();
    }
    if (kt + 2 < nkt)                     // hint tile kt+2 (global_prefetch_b8)
      __builtin_prefetch(bP0, 0, 3);

    // ---- fragment gathers: two aligned b128 LDS loads each ----
    AFrag af[2], bf[4];
#pragma unroll
    for (int fm = 0; fm < 2; ++fm) {
      const unsigned int* p =
          &lAp[buf][(waveM * 32 + fm * 16 + r16) * PSTR];
      af[fm].q[0] = *(const uint4*)(p + hi * 4);        // K pairs 0..3 (+4*hi)
      af[fm].q[1] = *(const uint4*)(p + 8 + hi * 4);    // K pairs 8..11 (+4*hi)
    }
#pragma unroll
    for (int fn = 0; fn < 4; ++fn) {
      const unsigned int* p =
          &lBp[buf][(waveN * 64 + fn * 16 + r16) * PSTR];
      bf[fn].q[0] = *(const uint4*)(p + hi * 4);        // K pairs 0..3 (+4*hi)
      bf[fn].q[1] = *(const uint4*)(p + 8 + hi * 4);    // K pairs 8..11 (+4*hi)
    }
#pragma unroll
    for (int fm = 0; fm < 2; ++fm)
#pragma unroll
      for (int fn = 0; fn < 4; ++fn)
        acc[fm][fn] = __builtin_amdgcn_wmma_f32_16x16x32_bf16(
            false, af[fm].v, false, bf[fn].v, (short)0, acc[fm][fn],
            false, false);

    if (more) {
      lstoreB(buf ^ 1);
      if (wave == 0) __builtin_amdgcn_s_wait_tensorcnt(0);
      __syncthreads();
    }
  }

  // ---- epilogue (C/D layout: VGPR j -> M=j+8*hi, N=lane%16) ----
#pragma unroll
  for (int fm = 0; fm < 2; ++fm) {
#pragma unroll
    for (int fn = 0; fn < 4; ++fn) {
      int ncol = nBase + waveN * 64 + fn * 16 + r16;
      float bv = bias[ncol];
#pragma unroll
      for (int j = 0; j < 8; ++j) {
        int mrow = mBase + waveM * 32 + fm * 16 + j + hi * 8;
        float val = acc[fm][fn][j] + bv;
        if (EPI == 0) {
          ((unsigned short*)Cvoid)[(long long)mrow * ldc + ncol] =
              f2bf(fmaxf(val, 0.f));
        } else {
          float s = scale[(long long)mrow * sstride];
          float* op = (float*)Cvoid + (long long)mrow * ldc + ncol;
          *op += s * val;
        }
      }
    }
  }
}

// ============================================================
// launch
// ============================================================
extern "C" void kernel_launch(void* const* d_in, const int* in_sizes, int n_in,
                              void* d_out, int out_size, void* d_ws,
                              size_t ws_size, hipStream_t stream) {
  const float* id_emb  = (const float*)d_in[0];
  const float* llm_emb = (const float*)d_in[1];
  const float* W1  = (const float*)d_in[2];
  const float* b1  = (const float*)d_in[3];
  const float* W2  = (const float*)d_in[4];
  const float* b2  = (const float*)d_in[5];
  const float* gW1 = (const float*)d_in[6];
  const float* gb1 = (const float*)d_in[7];
  const float* gW2 = (const float*)d_in[8];
  const float* gb2 = (const float*)d_in[9];
  float* out = (float*)d_out;
  const int B = in_sizes[0] / ID_DIM;   // 8192

  // workspace carve-up (256B aligned); ~210 MB total
  char* ws = (char*)d_ws;
  size_t off = 0;
  auto carve = [&](size_t bytes) {
    char* p = ws + off;
    off = (off + bytes + 255) & ~(size_t)255;
    return p;
  };
  unsigned short* combined = (unsigned short*)carve((size_t)B * IN_DIM * 2);
  unsigned short* hbuf     = (unsigned short*)carve((size_t)B * HID * 2);
  unsigned short* w1e      = (unsigned short*)carve((size_t)IN_DIM * HID * 2);
  unsigned short* w2e      = (unsigned short*)carve((size_t)HID * OUT_DIM * 2);
  float*          dense_w  = (float*)carve((size_t)B * NEXP * 4);
  (void)ws_size; (void)n_in; (void)out_size;

  // 1) zero output (accumulated across experts)
  long long out4 = (long long)B * OUT_DIM / 4;
  zero_f32_k<<<(unsigned)((out4 + 255) / 256), 256, 0, stream>>>(out, out4);

  // 2) pack concat(id,llm) -> bf16
  long long pk = (long long)B * (IN_DIM / 4);
  pack_combined_k<<<(unsigned)((pk + 255) / 256), 256, 0, stream>>>(
      id_emb, llm_emb, combined, B);

  // 3) gate -> dense per-expert weights
  gate_k<<<(B + 3) / 4, 128, 0, stream>>>(id_emb, llm_emb, gW1, gb1, gW2, gb2,
                                          dense_w, B);

  // 4) dense experts: convert weights, GEMM1 (relu), GEMM2 (scaled accumulate)
  dim3 g1(HID / TN, B / TM);      // 32 x 64
  dim3 g2(OUT_DIM / TN, B / TM);  // 16 x 64
  const long long n1 = (long long)IN_DIM * HID;   // 20.97M
  const long long n2 = (long long)HID * OUT_DIM;  // 8.39M
  for (int e = 0; e < NEXP; ++e) {
    cvt_bf16_k<<<(unsigned)((n1 / 4 + 255) / 256), 256, 0, stream>>>(
        W1 + (long long)e * n1, w1e, n1 / 4);
    cvt_bf16_k<<<(unsigned)((n2 / 4 + 255) / 256), 256, 0, stream>>>(
        W2 + (long long)e * n2, w2e, n2 / 4);
    gemm_bf16_wmma<0><<<g1, 256, 0, stream>>>(
        combined, IN_DIM, w1e, HID, b1 + (long long)e * HID,
        (const float*)nullptr, 0, (void*)hbuf, HID, IN_DIM);
    gemm_bf16_wmma<1><<<g2, 256, 0, stream>>>(
        hbuf, HID, w2e, OUT_DIM, b2 + (long long)e * OUT_DIM,
        dense_w + e, NEXP, (void*)out, OUT_DIM, HID);
  }
}
